// QueryPlanGNN_58334245814498
// MI455X (gfx1250) — matrix operand order repaired and checked by reading, not compiled, not run on hardware
//
#include <hip/hip_runtime.h>
#include <hip/hip_bf16.h>
#include <stdint.h>

// ---------- WMMA vector types (gfx1250, wave32) ----------
typedef __attribute__((ext_vector_type(16))) __bf16 v16bf;
typedef __attribute__((ext_vector_type(8)))  float  v8f;

#define H_DIM 128
#define BLOCK 256

// float -> bf16 (round to nearest even), stored as u16
__device__ __forceinline__ unsigned short f2bf(float f) {
    unsigned int u = __float_as_uint(f);
    unsigned int r = u + 0x7FFFu + ((u >> 16) & 1u);
    return (unsigned short)(r >> 16);
}

// ---------------- degree / norm ----------------
__global__ void k_deg_init(float* deg, float* g, int n) {
    int i = blockIdx.x * blockDim.x + threadIdx.x;
    if (i < n) deg[i] = 1.0f;          // self-loop
    if (i < H_DIM) g[i] = 0.0f;        // zero pooled accumulator
}

__global__ void k_deg_count(float* deg, const int* dst, int e) {
    int i = blockIdx.x * blockDim.x + threadIdx.x;
    if (i < e) unsafeAtomicAdd(&deg[dst[i]], 1.0f);
}

__global__ void k_dinv(float* deg, int n) {
    int i = blockIdx.x * blockDim.x + threadIdx.x;
    if (i < n) deg[i] = rsqrtf(fmaxf(deg[i], 1.0f));
}

// ---------------- weight pre-pack into WMMA B-fragment layout ----------------
// B fragment (16-bit, 32x16 K x N tile): lanes 0-15 hold N=lane, K=0..7,16..23;
// lanes 16-31 hold N=lane-16, K=8..15,24..31 (elem i -> k=(i&7)+16*(i>>3)+8*half).
// wpack layout: [layer][kt(4)][jt(8)][lane(32)][i(16)] bf16
__global__ void k_pack_w(const float* W1, const float* W2, const float* W3,
                         unsigned short* wpack) {
    int t = blockIdx.x * blockDim.x + threadIdx.x;      // 3*16384 threads
    if (t >= 3 * H_DIM * H_DIM) return;
    int w   = t >> 14;            // which weight matrix
    int rem = t & 16383;
    int i    = rem & 15;
    int lane = (rem >> 4) & 31;
    int jt   = (rem >> 9) & 7;
    int kt   = rem >> 12;
    int nLoc = lane & 15;
    int kLoc = (i & 7) + ((i >> 3) << 4) + ((lane < 16) ? 0 : 8);
    int k = kt * 32 + kLoc;
    int n = jt * 16 + nLoc;
    const float* W = (w == 0) ? W1 : (w == 1) ? W2 : W3;
    wpack[t] = f2bf(W[k * H_DIM + n]);
}

// ---------------- cast activations to bf16 ----------------
__global__ void k_cast_bf16(const float* in, unsigned short* out, int total) {
    int i = blockIdx.x * blockDim.x + threadIdx.x;
    if (i < total) out[i] = f2bf(in[i]);
}

// ---------------- WMMA GEMM: hw[N x 128] = hb(bf16) @ Wpacked ----------------
// 256 threads = 8 waves; wave w computes rows [base + 16w, base + 16w + 16) x 128 cols.
// Packed weights (32 KB) are staged into LDS once per block via the CDNA5 async
// global->LDS copy path (ASYNCcnt), then WMMA B fragments come from ds_load_b128.
__global__ __launch_bounds__(BLOCK) void k_gemm_wmma(
        const unsigned short* __restrict__ hb,
        const unsigned short* __restrict__ wpack,
        float* __restrict__ hw, int nRows) {
    __shared__ unsigned short wLDS[H_DIM * H_DIM];      // 32 KB packed weights

    // ---- async stage: 256 threads x 8 iters x 16B = 32768 B ----
    {
        const uint32_t ldsBase = (uint32_t)(uintptr_t)wLDS;   // low 32b = LDS offset
        const char* gbase = (const char*)wpack;
        const uint32_t t16 = (uint32_t)threadIdx.x * 16u;
#pragma unroll
        for (int it = 0; it < 8; ++it) {
            uint32_t off = t16 + (uint32_t)it * 4096u;
            uint32_t ldsAddr = ldsBase + off;
            uint64_t gaddr = (uint64_t)(uintptr_t)(gbase + off);
            asm volatile("global_load_async_to_lds_b128 %0, %1, off"
                         :: "v"(ldsAddr), "v"(gaddr) : "memory");
        }
        asm volatile("s_wait_asynccnt 0x0" ::: "memory");
    }
    __syncthreads();

    const int wave = threadIdx.x >> 5;
    const int lane = threadIdx.x & 31;
    const int rowBase = blockIdx.x * 128 + wave * 16;

    const int m = lane & 15;
    int rowA = rowBase + m;
    if (rowA > nRows - 1) rowA = nRows - 1;     // clamp (stores are guarded)
    const int kShift = (lane < 16) ? 0 : 8;     // A layout: hi-half lanes carry K+8

    v8f acc[8];
#pragma unroll
    for (int j = 0; j < 8; ++j) acc[j] = (v8f){0.f,0.f,0.f,0.f,0.f,0.f,0.f,0.f};

    const unsigned short* arow = hb + (size_t)rowA * H_DIM;

#pragma unroll
    for (int kt = 0; kt < 4; ++kt) {
        const int k0 = kt * 32 + kShift;
        union { uint4 q[2]; v16bf v; } A;
        A.q[0] = *(const uint4*)(arow + k0);        // K = k0 .. k0+7
        A.q[1] = *(const uint4*)(arow + k0 + 16);   // K = k0+16 .. k0+23
#pragma unroll
        for (int jt = 0; jt < 8; ++jt) {
            union { uint4 q[2]; v16bf v; } B;
            const uint4* wp = (const uint4*)(wLDS + (((kt * 8 + jt) * 32 + lane) << 4));
            B.q[0] = wp[0];
            B.q[1] = wp[1];
            acc[jt] = __builtin_amdgcn_wmma_f32_16x16x32_bf16(
                false, A.v, false, B.v, (short)0, acc[jt], false, false);
        }
    }

    // C/D layout: VGPR r: lanes 0-15 -> M=r, N=lane; lanes 16-31 -> M=r+8, N=lane-16
    const int n0   = lane & 15;
    const int rOff = (lane < 16) ? 0 : 8;
#pragma unroll
    for (int jt = 0; jt < 8; ++jt) {
#pragma unroll
        for (int r = 0; r < 8; ++r) {
            int row = rowBase + rOff + r;
            if (row < nRows) hw[(size_t)row * H_DIM + jt * 16 + n0] = acc[jt][r];
        }
    }
}

// ---------------- self-loop: agg[n] = dinv[n]^2 * hw[n] ----------------
__global__ void k_selfloop(const float* __restrict__ hw, const float* __restrict__ dinv,
                           float* __restrict__ agg, int n) {
    int t = blockIdx.x * blockDim.x + threadIdx.x;   // n*32 threads, 4 floats each
    int node = t >> 5;
    if (node >= n) return;
    int f = (t & 31) * 4;
    float d = dinv[node];
    float c = d * d;
    const float4 v = *(const float4*)(hw + (size_t)node * H_DIM + f);
    float4 o; o.x = c * v.x; o.y = c * v.y; o.z = c * v.z; o.w = c * v.w;
    *(float4*)(agg + (size_t)node * H_DIM + f) = o;
}

// ---------------- edge scatter: agg[dst] += dinv[s]*dinv[d] * hw[src] ----------------
__global__ void k_edge(const float* __restrict__ hw, const float* __restrict__ dinv,
                       const int* __restrict__ src, const int* __restrict__ dst,
                       float* __restrict__ agg, int e) {
    int t = blockIdx.x * blockDim.x + threadIdx.x;   // e*32 threads, 4 floats each
    int edge = t >> 5;
    if (edge >= e) return;
    int f = (t & 31) * 4;
    int s = src[edge];
    int d = dst[edge];
    float c = dinv[s] * dinv[d];
    const float4 v = *(const float4*)(hw + (size_t)s * H_DIM + f);
    float* a = agg + (size_t)d * H_DIM + f;
    unsafeAtomicAdd(a + 0, c * v.x);
    unsafeAtomicAdd(a + 1, c * v.y);
    unsafeAtomicAdd(a + 2, c * v.z);
    unsafeAtomicAdd(a + 3, c * v.w);
}

// ---------------- finalize: +bias, ReLU, requantize to bf16 for next layer ----------------
__global__ void k_finalize_relu_bf16(const float* __restrict__ agg, const float* __restrict__ b,
                                     unsigned short* __restrict__ hb, int total) {
    int t = blockIdx.x * blockDim.x + threadIdx.x;
    if (t >= total) return;
    float v = agg[t] + b[t & (H_DIM - 1)];
    hb[t] = f2bf(fmaxf(v, 0.0f));
}

// layer 3: +bias only, keep f32 (written into hw buffer, which is dead by now)
__global__ void k_finalize_f32(const float* __restrict__ agg, const float* __restrict__ b,
                               float* __restrict__ out, int total) {
    int t = blockIdx.x * blockDim.x + threadIdx.x;
    if (t >= total) return;
    out[t] = agg[t] + b[t & (H_DIM - 1)];
}

// ---------------- global mean pool (partials into g via atomics) ----------------
__global__ void k_pool(const float* __restrict__ h3, float* __restrict__ g, int n) {
    int k = threadIdx.x & (H_DIM - 1);
    int r0 = blockIdx.x * 2 + (threadIdx.x >> 7);
    int stride = gridDim.x * 2;
    float s = 0.0f;
    for (int r = r0; r < n; r += stride) s += h3[(size_t)r * H_DIM + k];
    unsafeAtomicAdd(&g[k], s);
}

// ---------------- head: relu(mean @ Wp + bp) @ Wc + bc ----------------
__global__ void k_head(const float* __restrict__ g, const float* __restrict__ Wp,
                       const float* __restrict__ bp, const float* __restrict__ Wc,
                       const float* __restrict__ bc, float* __restrict__ out, int n) {
    __shared__ float gm[H_DIM];
    __shared__ float ps[H_DIM];
    int j = threadIdx.x;                 // 128 threads
    gm[j] = g[j] / (float)n;
    __syncthreads();
    float acc = bp[j];
#pragma unroll 8
    for (int k = 0; k < H_DIM; ++k) acc += gm[k] * Wp[k * H_DIM + j];
    ps[j] = fmaxf(acc, 0.0f) * Wc[j];
    __syncthreads();
    for (int s = 64; s > 0; s >>= 1) {
        if (j < s) ps[j] += ps[j + s];
        __syncthreads();
    }
    if (j == 0) out[0] = ps[0] + bc[0];
}

// ============================================================================
extern "C" void kernel_launch(void* const* d_in, const int* in_sizes, int n_in,
                              void* d_out, int out_size, void* d_ws, size_t ws_size,
                              hipStream_t stream) {
    const float* x   = (const float*)d_in[0];
    const int*   ei  = (const int*)  d_in[1];
    const float* W1  = (const float*)d_in[2];
    const float* b1  = (const float*)d_in[3];
    const float* W2  = (const float*)d_in[4];
    const float* b2  = (const float*)d_in[5];
    const float* W3  = (const float*)d_in[6];
    const float* b3  = (const float*)d_in[7];
    const float* Wp  = (const float*)d_in[8];
    const float* bp  = (const float*)d_in[9];
    const float* Wc  = (const float*)d_in[10];
    const float* bc  = (const float*)d_in[11];
    float* out = (float*)d_out;

    const int N = in_sizes[0] / H_DIM;      // 100000
    const int E = in_sizes[1] / 2;          // 1600000
    const int* src = ei;
    const int* dst = ei + E;

    // workspace carve-up (256B aligned)
    uintptr_t base = (uintptr_t)d_ws;
    size_t off = 0;
    auto carve = [&](size_t bytes) -> uintptr_t {
        uintptr_t p = base + off;
        off = (off + bytes + 255) & ~(size_t)255;
        return p;
    };
    float*          dinv  = (float*)         carve((size_t)N * 4);
    unsigned short* hb    = (unsigned short*)carve((size_t)N * H_DIM * 2);
    float*          hw    = (float*)         carve((size_t)N * H_DIM * 4);
    float*          agg   = (float*)         carve((size_t)N * H_DIM * 4);
    unsigned short* wpack = (unsigned short*)carve((size_t)3 * H_DIM * H_DIM * 2);
    float*          g     = (float*)         carve((size_t)H_DIM * 4);
    (void)ws_size;

    const int totNF   = N * H_DIM;
    const int gemmGrid = (N + 127) / 128;

    // degree / symmetric norm
    k_deg_init <<<(N + BLOCK - 1) / BLOCK, BLOCK, 0, stream>>>(dinv, g, N);
    k_deg_count<<<(E + BLOCK - 1) / BLOCK, BLOCK, 0, stream>>>(dinv, dst, E);
    k_dinv     <<<(N + BLOCK - 1) / BLOCK, BLOCK, 0, stream>>>(dinv, N);

    // pack all three weight matrices into WMMA B-fragment layout (bf16)
    k_pack_w<<<(3 * H_DIM * H_DIM + BLOCK - 1) / BLOCK, BLOCK, 0, stream>>>(W1, W2, W3, wpack);

    // layer-1 input activations -> bf16
    k_cast_bf16<<<(totNF + BLOCK - 1) / BLOCK, BLOCK, 0, stream>>>(x, hb, totNF);

    const float* biases[3] = { b1, b2, b3 };
    for (int layer = 0; layer < 3; ++layer) {
        const unsigned short* wl = wpack + (size_t)layer * H_DIM * H_DIM;
        k_gemm_wmma<<<gemmGrid, BLOCK, 0, stream>>>(hb, wl, hw, N);
        k_selfloop <<<((size_t)N * 32 + BLOCK - 1) / BLOCK, BLOCK, 0, stream>>>(hw, dinv, agg, N);
        k_edge     <<<((size_t)E * 32 + BLOCK - 1) / BLOCK, BLOCK, 0, stream>>>(hw, dinv, src, dst, agg, E);
        if (layer < 2) {
            k_finalize_relu_bf16<<<(totNF + BLOCK - 1) / BLOCK, BLOCK, 0, stream>>>(agg, biases[layer], hb, totNF);
        } else {
            k_finalize_f32<<<(totNF + BLOCK - 1) / BLOCK, BLOCK, 0, stream>>>(agg, biases[layer], hw, totNF);
        }
    }

    // global mean pool + MLP head
    k_pool<<<512, BLOCK, 0, stream>>>(hw, g, N);
    k_head<<<1, H_DIM, 0, stream>>>(g, Wp, bp, Wc, bc, out, N);
}
